// MambaBlock_38319698215457
// MI455X (gfx1250) — compile-verified, matrix operand order checked by reading, not run on hardware
//
#include <hip/hip_runtime.h>

#define B_ 8
#define L_ 2048
#define DIM_ 768
#define DIN_ 1536
#define DSTATE_ 16
#define DTRANK_ 48
#define DTPAD_ 64
#define NROWS_ (B_ * L_)          // 16384
#define XPN_ 80                   // DT_RANK + 2*D_STATE

typedef unsigned short BF;        // bf16 storage
typedef __bf16 bf16;
typedef __attribute__((ext_vector_type(8)))  __bf16 v8bf;
typedef __attribute__((ext_vector_type(16))) __bf16 v16bf;
typedef __attribute__((ext_vector_type(8)))  float  v8f;

__device__ __forceinline__ BF f2bf(float f) {
  unsigned u = __float_as_uint(f);
  unsigned r = u + 0x7FFFu + ((u >> 16) & 1u);   // RNE
  return (BF)(r >> 16);
}
__device__ __forceinline__ float sigf(float v) { return 1.0f / (1.0f + __expf(-v)); }
__device__ __forceinline__ float softplusf(float v) { return (v > 20.0f) ? v : log1pf(__expf(v)); }

// Load a 16-lane-row bf16 fragment: two contiguous 8-elem (16B) K-chunks at +0 and +16
// (caller pre-offsets pointer by row*K + k0 + half*8 per ISA 7.12.2 16-bit A/B layout).
__device__ __forceinline__ v16bf ld_frag(const bf16* p) {
  v8bf lo = *(const v8bf*)p;
  v8bf hi = *(const v8bf*)(p + 16);
  v16bf r;
#pragma unroll
  for (int i = 0; i < 8; ++i) { r[i] = lo[i]; r[i + 8] = hi[i]; }
  return r;
}

// One wave computes a 16(M) x 64(N) strip: 4 N-tiles sharing the A fragment.
// Software-pipelined: next K-step's 10 b128 loads are issued before this step's
// 4 WMMAs so loads overlap matrix math instead of wait->wmma serialization.
// All guards (more, bv[j]) are wave-uniform -> EXEC stays all-ones for WMMA.
__device__ __forceinline__ void gemm16x64(const bf16* __restrict__ A,
                                          const bf16* __restrict__ W,
                                          int K, int N, int m0, int n0b,
                                          int lane, v8f acc[4]) {
  const int idx = lane & 15, hh = lane >> 4;
  const bf16* arow = A + (size_t)(m0 + idx) * K + hh * 8;
  const bf16* brow[4];
  bool bv[4];
#pragma unroll
  for (int j = 0; j < 4; ++j) {
    int n0 = n0b + j * 16;
    bv[j] = (n0 < N);
    brow[j] = W + (size_t)((bv[j] ? n0 : n0b) + idx) * K + hh * 8;
  }
  v16bf af = ld_frag(arow);
  v16bf bfr[4];
#pragma unroll
  for (int j = 0; j < 4; ++j) if (bv[j]) bfr[j] = ld_frag(brow[j]);

  for (int k0 = 32; k0 < K + 32; k0 += 32) {
    const bool more = (k0 < K);
    v16bf afn = af;
    v16bf bfn[4];
    if (more) {
      __builtin_prefetch(arow + k0 + 32, 0, 1);      // global_prefetch_b8
      afn = ld_frag(arow + k0);
#pragma unroll
      for (int j = 0; j < 4; ++j) if (bv[j]) bfn[j] = ld_frag(brow[j] + k0);
    }
#pragma unroll
    for (int j = 0; j < 4; ++j)
      if (bv[j])
        acc[j] = __builtin_amdgcn_wmma_f32_16x16x32_bf16(false, af, false, bfr[j],
                                                         (short)0, acc[j], false, false);
    if (more) {
      af = afn;
#pragma unroll
      for (int j = 0; j < 4; ++j) if (bv[j]) bfr[j] = bfn[j];
    }
  }
}

// ---------- prep ----------
__global__ void k_cvt(const float* __restrict__ s, BF* __restrict__ d, int n) {
  int i = blockIdx.x * blockDim.x + threadIdx.x;
  if (i < n) d[i] = f2bf(s[i]);
}
__global__ void k_prep_wdt(const float* __restrict__ wdt, BF* __restrict__ d) {
  int i = blockIdx.x * blockDim.x + threadIdx.x;
  if (i >= DIN_ * DTPAD_) return;
  int r = i / DTPAD_, c = i % DTPAD_;
  d[i] = (c < DTRANK_) ? f2bf(wdt[r * DTRANK_ + c]) : (BF)0;
}
__global__ void k_prep_A(const float* __restrict__ alog, float* __restrict__ an) {
  int i = blockIdx.x * blockDim.x + threadIdx.x;
  if (i < DIN_ * DSTATE_) an[i] = -__expf(alog[i]);
}

// ---------- layernorm + mask -> bf16 ----------
__global__ void k_ln(const float* __restrict__ x, const int* __restrict__ mask,
                     const float* __restrict__ g, const float* __restrict__ b,
                     BF* __restrict__ xn) {
  __shared__ float red[256];
  const int r = blockIdx.x, tid = threadIdx.x;
  const float* row = x + (size_t)r * DIM_;
  float v[3], s = 0.f, ss = 0.f;
#pragma unroll
  for (int i = 0; i < 3; ++i) { v[i] = row[tid + i * 256]; s += v[i]; ss += v[i] * v[i]; }
  red[tid] = s; __syncthreads();
  for (int o = 128; o > 0; o >>= 1) { if (tid < o) red[tid] += red[tid + o]; __syncthreads(); }
  float mean = red[0] / (float)DIM_; __syncthreads();
  red[tid] = ss; __syncthreads();
  for (int o = 128; o > 0; o >>= 1) { if (tid < o) red[tid] += red[tid + o]; __syncthreads(); }
  float var = red[0] / (float)DIM_ - mean * mean;
  float inv = rsqrtf(var + 1e-5f);
  float mf = (float)mask[r];
#pragma unroll
  for (int i = 0; i < 3; ++i) {
    int c = tid + i * 256;
    xn[(size_t)r * DIM_ + c] = f2bf(((v[i] - mean) * inv * g[c] + b[c]) * mf);
  }
}

// ---------- GEMM1: xz = xn @ W_in^T ; split x_in / silu(z) ----------
__global__ void k_gemm_inproj(const BF* __restrict__ A, const BF* __restrict__ W,
                              float* __restrict__ xin, float* __restrict__ sz) {
  const int NT = (2 * DIN_) / 64;                 // 48
  int wid = (blockIdx.x * blockDim.x + threadIdx.x) >> 5;
  int lane = threadIdx.x & 31;
  if (wid >= (NROWS_ / 16) * NT) return;
  int m0 = (wid / NT) * 16, n0b = (wid % NT) * 64;
  v8f acc[4] = {};
  gemm16x64((const bf16*)A, (const bf16*)W, DIM_, 2 * DIN_, m0, n0b, lane, acc);
  int idx = lane & 15, hh = lane >> 4;
#pragma unroll
  for (int j = 0; j < 4; ++j) {
    int n0 = n0b + j * 16;
#pragma unroll
    for (int v = 0; v < 8; ++v) {
      int row = m0 + hh * 8 + v, col = n0 + idx;
      float a = acc[j][v];
      if (col < DIN_) xin[(size_t)row * DIN_ + col] = a;
      else            sz[(size_t)row * DIN_ + (col - DIN_)] = a * sigf(a);
    }
  }
}

// ---------- depthwise causal conv (k=4) + SiLU ----------
__global__ void k_conv(const float* __restrict__ xin, const float* __restrict__ cw,
                       const float* __restrict__ cb, float* __restrict__ xcf,
                       BF* __restrict__ xcb) {
  int i = blockIdx.x * blockDim.x + threadIdx.x;
  if (i >= NROWS_ * DIN_) return;
  int d = i % DIN_, m = i / DIN_, l = m & (L_ - 1);
  float acc = cb[d];
#pragma unroll
  for (int j = 0; j < 4; ++j) {
    int ll = l - 3 + j;
    if (ll >= 0) acc += cw[d * 4 + j] * xin[(size_t)(m - 3 + j) * DIN_ + d];
  }
  float sv = acc * sigf(acc);
  xcf[i] = sv; xcb[i] = f2bf(sv);
}

// ---------- GEMM2: xdbl = xc @ W_xp^T (N=80) ; split dt_in / B / C ----------
__global__ void k_gemm_xproj(const BF* __restrict__ A, const BF* __restrict__ W,
                             BF* __restrict__ dtin, float* __restrict__ Bm,
                             float* __restrict__ Cm) {
  const int NT = 2;                               // ceil(80/64)
  int wid = (blockIdx.x * blockDim.x + threadIdx.x) >> 5;
  int lane = threadIdx.x & 31;
  if (wid >= (NROWS_ / 16) * NT) return;
  int m0 = (wid / NT) * 16, n0b = (wid % NT) * 64;
  v8f acc[4] = {};
  gemm16x64((const bf16*)A, (const bf16*)W, DIN_, XPN_, m0, n0b, lane, acc);
  int idx = lane & 15, hh = lane >> 4;
#pragma unroll
  for (int j = 0; j < 4; ++j) {
    int n0 = n0b + j * 16;
    if (n0 >= XPN_) continue;
#pragma unroll
    for (int v = 0; v < 8; ++v) {
      int row = m0 + hh * 8 + v, col = n0 + idx;
      float a = acc[j][v];
      if (col < DTRANK_)                 dtin[(size_t)row * DTPAD_ + col] = f2bf(a);
      else if (col < DTRANK_ + DSTATE_)  Bm[(size_t)row * DSTATE_ + (col - DTRANK_)] = a;
      else                               Cm[(size_t)row * DSTATE_ + (col - DTRANK_ - DSTATE_)] = a;
    }
  }
}

// ---------- GEMM3: dt = softplus(dt_in @ W_dt^T + b_dt) ----------
__global__ void k_gemm_dt(const BF* __restrict__ A, const BF* __restrict__ W,
                          const float* __restrict__ bdt, float* __restrict__ dt) {
  const int NT = DIN_ / 64;                       // 24
  int wid = (blockIdx.x * blockDim.x + threadIdx.x) >> 5;
  int lane = threadIdx.x & 31;
  if (wid >= (NROWS_ / 16) * NT) return;
  int m0 = (wid / NT) * 16, n0b = (wid % NT) * 64;
  v8f acc[4] = {};
  gemm16x64((const bf16*)A, (const bf16*)W, DTPAD_, DIN_, m0, n0b, lane, acc);
  int idx = lane & 15, hh = lane >> 4;
#pragma unroll
  for (int j = 0; j < 4; ++j) {
    int n0 = n0b + j * 16;
#pragma unroll
    for (int v = 0; v < 8; ++v) {
      int row = m0 + hh * 8 + v, col = n0 + idx;
      dt[(size_t)row * DIN_ + col] = softplusf(acc[j][v] + bdt[col]);
    }
  }
}

// ---------- selective scan: h = exp(dt*A)h + dt*x*B ; y = <h,C> ; gate+skip ----------
__global__ void k_scan(const float* __restrict__ dtb, const float* __restrict__ xcf,
                       const float* __restrict__ Bmb, const float* __restrict__ Cmb,
                       const float* __restrict__ An, const float* __restrict__ Dp,
                       const float* __restrict__ szf, BF* __restrict__ yg) {
  __shared__ float sB[32 * DSTATE_], sC[32 * DSTATE_];
  const int blkPerB = DIN_ / 256;                 // 6
  const int b = blockIdx.x / blkPerB;
  const int d = (blockIdx.x % blkPerB) * 256 + threadIdx.x;
  float h[DSTATE_], Ar[DSTATE_];
#pragma unroll
  for (int s = 0; s < DSTATE_; ++s) { h[s] = 0.f; Ar[s] = An[d * DSTATE_ + s]; }
  const float dval = Dp[d];
  for (int tb = 0; tb < L_; tb += 32) {
    for (int i = threadIdx.x; i < 32 * DSTATE_; i += 256) {
      int t = i >> 4, s = i & 15;
      size_t m = (size_t)b * L_ + tb + t;
      sB[i] = Bmb[m * DSTATE_ + s];
      sC[i] = Cmb[m * DSTATE_ + s];
    }
    __syncthreads();
    for (int t = 0; t < 32; ++t) {
      size_t m = (size_t)b * L_ + tb + t;
      float dt = dtb[m * DIN_ + d];
      float xv = xcf[m * DIN_ + d];
      float dtx = dt * xv, y = 0.f;
#pragma unroll
      for (int s = 0; s < DSTATE_; ++s) {
        float e = __expf(dt * Ar[s]);
        h[s] = e * h[s] + dtx * sB[t * DSTATE_ + s];
        y += h[s] * sC[t * DSTATE_ + s];
      }
      yg[m * DIN_ + d] = f2bf((y + xv * dval) * szf[m * DIN_ + d]);
    }
    __syncthreads();
  }
}

// ---------- GEMM4: out = (x + ygate @ W_out^T) * mask ----------
__global__ void k_gemm_out(const BF* __restrict__ A, const BF* __restrict__ W,
                           const float* __restrict__ x, const int* __restrict__ mask,
                           float* __restrict__ out) {
  const int NT = DIM_ / 64;                       // 12
  int wid = (blockIdx.x * blockDim.x + threadIdx.x) >> 5;
  int lane = threadIdx.x & 31;
  if (wid >= (NROWS_ / 16) * NT) return;
  int m0 = (wid / NT) * 16, n0b = (wid % NT) * 64;
  v8f acc[4] = {};
  gemm16x64((const bf16*)A, (const bf16*)W, DIN_, DIM_, m0, n0b, lane, acc);
  int idx = lane & 15, hh = lane >> 4;
#pragma unroll
  for (int j = 0; j < 4; ++j) {
    int n0 = n0b + j * 16;
#pragma unroll
    for (int v = 0; v < 8; ++v) {
      int row = m0 + hh * 8 + v, col = n0 + idx;
      float mf = (float)mask[row];
      out[(size_t)row * DIM_ + col] = (x[(size_t)row * DIM_ + col] + acc[j][v]) * mf;
    }
  }
}

extern "C" void kernel_launch(void* const* d_in, const int* in_sizes, int n_in,
                              void* d_out, int out_size, void* d_ws, size_t ws_size,
                              hipStream_t stream) {
  const float* x      = (const float*)d_in[0];
  const int*   mask   = (const int*)  d_in[1];
  const float* ln_g   = (const float*)d_in[2];
  const float* ln_b   = (const float*)d_in[3];
  const float* W_in   = (const float*)d_in[4];
  const float* conv_w = (const float*)d_in[5];
  const float* conv_b = (const float*)d_in[6];
  const float* W_xp   = (const float*)d_in[7];
  const float* W_dt   = (const float*)d_in[8];
  const float* b_dt   = (const float*)d_in[9];
  const float* A_log  = (const float*)d_in[10];
  const float* Dp     = (const float*)d_in[11];
  const float* W_out  = (const float*)d_in[12];
  float* out = (float*)d_out;

  char* ws = (char*)d_ws; size_t off = 0;
  auto alloc = [&](size_t bytes) { void* p = ws + off; off = (off + bytes + 255) & ~(size_t)255; return p; };
  BF*    xn_bf   = (BF*)   alloc((size_t)NROWS_ * DIM_ * 2);
  BF*    Win_bf  = (BF*)   alloc((size_t)2 * DIN_ * DIM_ * 2);
  BF*    Wxp_bf  = (BF*)   alloc((size_t)XPN_ * DIN_ * 2);
  BF*    Wdt_bf  = (BF*)   alloc((size_t)DIN_ * DTPAD_ * 2);
  BF*    Wout_bf = (BF*)   alloc((size_t)DIM_ * DIN_ * 2);
  float* An      = (float*)alloc((size_t)DIN_ * DSTATE_ * 4);
  float* xin_f   = (float*)alloc((size_t)NROWS_ * DIN_ * 4);
  float* sz_f    = (float*)alloc((size_t)NROWS_ * DIN_ * 4);
  float* xc_f    = (float*)alloc((size_t)NROWS_ * DIN_ * 4);
  BF*    xc_bf   = (BF*)   alloc((size_t)NROWS_ * DIN_ * 2);
  BF*    dtin_bf = (BF*)   alloc((size_t)NROWS_ * DTPAD_ * 2);
  float* Bm      = (float*)alloc((size_t)NROWS_ * DSTATE_ * 4);
  float* Cm      = (float*)alloc((size_t)NROWS_ * DSTATE_ * 4);
  float* dt_f    = (float*)alloc((size_t)NROWS_ * DIN_ * 4);
  BF*    yg_bf   = (BF*)   alloc((size_t)NROWS_ * DIN_ * 2);
  (void)ws_size; (void)n_in; (void)in_sizes; (void)out_size;

  auto cdiv = [](long a, long b) { return (int)((a + b - 1) / b); };

  // weight prep
  k_cvt<<<cdiv((long)2 * DIN_ * DIM_, 256), 256, 0, stream>>>(W_in, Win_bf, 2 * DIN_ * DIM_);
  k_cvt<<<cdiv((long)XPN_ * DIN_, 256), 256, 0, stream>>>(W_xp, Wxp_bf, XPN_ * DIN_);
  k_cvt<<<cdiv((long)DIM_ * DIN_, 256), 256, 0, stream>>>(W_out, Wout_bf, DIM_ * DIN_);
  k_prep_wdt<<<cdiv((long)DIN_ * DTPAD_, 256), 256, 0, stream>>>(W_dt, Wdt_bf);
  k_prep_A<<<cdiv((long)DIN_ * DSTATE_, 256), 256, 0, stream>>>(A_log, An);
  hipMemsetAsync(dtin_bf, 0, (size_t)NROWS_ * DTPAD_ * 2, stream);   // zero K-pad cols 48..63

  // pipeline
  k_ln<<<NROWS_, 256, 0, stream>>>(x, mask, ln_g, ln_b, xn_bf);

  { int waves = (NROWS_ / 16) * ((2 * DIN_) / 64);
    k_gemm_inproj<<<waves / 8, 256, 0, stream>>>(xn_bf, Win_bf, xin_f, sz_f); }

  k_conv<<<cdiv((long)NROWS_ * DIN_, 256), 256, 0, stream>>>(xin_f, conv_w, conv_b, xc_f, xc_bf);

  { int waves = (NROWS_ / 16) * 2;
    k_gemm_xproj<<<waves / 8, 256, 0, stream>>>(xc_bf, Wxp_bf, dtin_bf, Bm, Cm); }

  { int waves = (NROWS_ / 16) * (DIN_ / 64);
    k_gemm_dt<<<waves / 8, 256, 0, stream>>>(dtin_bf, Wdt_bf, b_dt, dt_f); }

  k_scan<<<B_ * (DIN_ / 256), 256, 0, stream>>>(dt_f, xc_f, Bm, Cm, An, Dp, sz_f, yg_bf);

  { int waves = (NROWS_ / 16) * (DIM_ / 64);
    k_gemm_out<<<waves / 8, 256, 0, stream>>>(yg_bf, Wout_bf, x, mask, out); }
}
